// LCONV_34179349741641
// MI455X (gfx1250) — matrix operand order repaired and checked by reading, not compiled, not run on hardware
//
#include <hip/hip_runtime.h>

typedef __attribute__((ext_vector_type(16))) __bf16 v16bf;
typedef __attribute__((ext_vector_type(8)))  float  v8f;
typedef unsigned int uint;

namespace {
constexpr int C_IN  = 128;
constexpr int HW    = 56;
constexpr int L_PIX = HW * HW;          // 3136
constexpr int BATCH = 32;
constexpr int F_OUT = 256;
constexpr int D_K   = C_IN * 9;         // 1152
constexpr int M_TOT = BATCH * L_PIX;    // 100352
constexpr int BM = 128;                 // pixels per block
constexpr int XPITCH = 160;             // bytes per xstage row (64ch*2B + 32B pad)
constexpr int WSTRIDE_R = F_OUT * C_IN * 2;   // 65536 B per tap in Wp
}

// ---------------------------------------------------------------------------
// Kernel 1: per-pixel channel sums  s1 = sum_c x, s2 = sum_c x^2
// ---------------------------------------------------------------------------
__global__ void k_prep_x(const float* __restrict__ x,
                         float* __restrict__ s1, float* __restrict__ s2) {
    int idx = blockIdx.x * 256 + threadIdx.x;
    if (idx >= M_TOT) return;
    int b = idx / L_PIX;
    int p = idx - b * L_PIX;
    const float* xp = x + (size_t)b * C_IN * L_PIX + p;
    float a1 = 0.f, a2 = 0.f;
#pragma unroll 4
    for (int c = 0; c < C_IN; ++c) {
        float v = xp[(size_t)c * L_PIX];
        a1 += v;
        a2 += v * v;
    }
    s1[idx] = a1;
    s2[idx] = a2;
}

// ---------------------------------------------------------------------------
// Kernel 2: 3x3 box filter (zero pad) of s1/s2 -> mean, inv_std per patch
// ---------------------------------------------------------------------------
__global__ void k_stats(const float* __restrict__ s1, const float* __restrict__ s2,
                        float* __restrict__ meanArr, float* __restrict__ invstdArr) {
    int idx = blockIdx.x * 256 + threadIdx.x;
    if (idx >= M_TOT) return;
    int b = idx / L_PIX;
    int p = idx - b * L_PIX;
    int h = p / HW, w = p - h * HW;
    float S1 = 0.f, S2 = 0.f;
#pragma unroll
    for (int dh = -1; dh <= 1; ++dh) {
#pragma unroll
        for (int dw = -1; dw <= 1; ++dw) {
            int hh = h + dh, ww = w + dw;
            if (hh >= 0 && hh < HW && ww >= 0 && ww < HW) {
                int q = b * L_PIX + hh * HW + ww;
                S1 += s1[q];
                S2 += s2[q];
            }
        }
    }
    const float invD = 1.0f / (float)D_K;
    float mean = S1 * invD;
    float var  = fmaxf(S2 * invD - mean * mean, 0.0f);
    meanArr[idx]   = mean;
    invstdArr[idx] = 1.0f / sqrtf(var + 1e-14f);
}

// ---------------------------------------------------------------------------
// Kernel 3: repack weights [F,C,3,3] -> bf16 Wp[r=kh*3+kw][F][C]; wsum[f]
// ---------------------------------------------------------------------------
__global__ void k_prep_w(const float* __restrict__ wsrc,
                         __bf16* __restrict__ Wp, float* __restrict__ wsum) {
    int f = blockIdx.x;          // 0..255
    int c = threadIdx.x;         // 0..127
    float s = 0.f;
#pragma unroll
    for (int r = 0; r < 9; ++r) {
        float v = wsrc[((size_t)f * C_IN + c) * 9 + r];   // (f,c,kh,kw), r = kh*3+kw
        s += v;
        Wp[((size_t)r * F_OUT + f) * C_IN + c] = (__bf16)v;
    }
    __shared__ float red[128];
    red[c] = s;
    __syncthreads();
#pragma unroll
    for (int off = 64; off > 0; off >>= 1) {
        if (c < off) red[c] += red[c + off];
        __syncthreads();
    }
    if (c == 0) wsum[f] = red[0];
}

// ---------------------------------------------------------------------------
// Kernel 4: im2col GEMM. x staged once per block in LDS as [pixel+halo][ch];
// taps = shifted LDS rows (zero-pad -> cndmask to a zeroed row, pre-scaled
// byte offsets so the hot loop is add+cndmask+loads+WMMA). B fragments come
// straight from L2-resident Wp in WMMA layout; no K-loop barriers.
// ---------------------------------------------------------------------------
__global__ __launch_bounds__(256) void k_gemm(
    const float* __restrict__ x, const __bf16* __restrict__ Wp,
    const float* __restrict__ meanArr, const float* __restrict__ invstdArr,
    const float* __restrict__ wsum, const float* __restrict__ bias,
    float* __restrict__ out) {

    // 257 rows (256 pixels + 1 zero row) x 160B pitch = 41120 B
    __shared__ __align__(16) char xs[257 * XPITCH];

    const int tid   = threadIdx.x;
    const int m0    = blockIdx.x * BM;
    const int gbase = m0 - 64;

    const int lane  = tid & 31;
    const int lm    = lane & 15;
    const int lh    = lane >> 4;
    const int waveId = tid >> 5;
    const int waveM = waveId & 1;        // 0..1 -> 64 rows
    const int waveN = waveId >> 1;       // 0..3 -> 64 cols

    // per-lane A-fragment rows (4 fragments along M), pre-scaled to bytes
    int rh[4], rw[4], rbyte[4];
#pragma unroll
    for (int i = 0; i < 4; ++i) {
        int m = m0 + waveM * 64 + i * 16 + lm;
        int b = m / L_PIX;
        int p = m - b * L_PIX;
        rh[i]    = p / HW;
        rw[i]    = p - rh[i] * HW;
        rbyte[i] = (m - gbase) * XPITCH + lh * 16;
    }
    const int zoff = 256 * XPITCH + lh * 16;     // zero-row redirect

    // staging mapping: one pixel row per thread
    const int srow = tid;
    int g  = gbase + srow;
    int gc = min(max(g, 0), M_TOT - 1);
    int sb = gc / L_PIX;
    int sp = gc - sb * L_PIX;

    // zero row (index 256)
    if (tid < XPITCH / 16) {
        uint4 z = {0u, 0u, 0u, 0u};
        *(uint4*)(xs + 256 * XPITCH + tid * 16) = z;
    }

    v8f acc[4][4];
#pragma unroll
    for (int i = 0; i < 4; ++i)
#pragma unroll
        for (int j = 0; j < 4; ++j)
#pragma unroll
            for (int q = 0; q < 8; ++q) acc[i][j][q] = 0.0f;

    union Frag { uint4 u[2]; v16bf v; };

    // per-lane part of the B pointer (WMMA B layout: lane lm = col,
    // half lh = K blocks of 8 at lh*8 and 16+lh*8)
    const char* wlane = (const char*)Wp + ((size_t)(waveN * 64 + lm) * C_IN + lh * 8) * 2;

#pragma unroll 1
    for (int cb = 0; cb < 2; ++cb) {
        __syncthreads();                 // previous phase fully consumed
        // ---- stage 64 channels of 256 pixels, transposed to [pix][ch]
        {
            const float* xp = x + ((size_t)sb * C_IN + cb * 64) * L_PIX + sp;
#pragma unroll
            for (int c8 = 0; c8 < 8; ++c8) {
                union { __bf16 h[8]; uint4 q; } pk;
#pragma unroll
                for (int e = 0; e < 8; ++e)
                    pk.h[e] = (__bf16)xp[(size_t)(c8 * 8 + e) * L_PIX];
                *(uint4*)(xs + srow * XPITCH + c8 * 16) = pk.q;
            }
        }
        __syncthreads();

        auto loadB = [&](const char* p, int ccOff, Frag bfr[4]) {
#pragma unroll
            for (int j = 0; j < 4; ++j) {
                bfr[j].u[0] = *(const uint4*)(p + ccOff + j * 4096);
                bfr[j].u[1] = *(const uint4*)(p + ccOff + j * 4096 + 32);
            }
        };

        auto computeChunk = [&](const int aoff[4], int ccOff, Frag bfr[4]) {
#pragma unroll
            for (int i = 0; i < 4; ++i) {
                Frag af;
                const char* pa = xs + aoff[i] + ccOff;
                af.u[0] = *(const uint4*)pa;          // K = cc + lh*8 .. +7
                af.u[1] = *(const uint4*)(pa + 32);   // K = cc + 16 + lh*8 ..
#pragma unroll
                for (int j = 0; j < 4; ++j)
                    acc[i][j] = __builtin_amdgcn_wmma_f32_16x16x32_bf16(
                        false, af.v, false, bfr[j].v, (short)0, acc[i][j],
                        false, false);
            }
        };

        // ---- K loop: 9 taps x 2 chunks of 32 channels, B one chunk ahead
        const char* wp = wlane + cb * 128;
        Frag bfA[4], bfB[4];
        loadB(wp, 0, bfA);
#pragma unroll 1
        for (int dh = -1; dh <= 1; ++dh) {
#pragma unroll 1
            for (int dw = -1; dw <= 1; ++dw) {
                const int doffB = (dh * HW + dw) * XPITCH;   // SALU
                int aoff[4];
#pragma unroll
                for (int i = 0; i < 4; ++i) {
                    bool v = ((unsigned)(rh[i] + dh) < (unsigned)HW) &
                             ((unsigned)(rw[i] + dw) < (unsigned)HW);
                    aoff[i] = v ? (rbyte[i] + doffB) : zoff;   // add + cndmask
                }
                loadB(wp, 64, bfB);
                computeChunk(aoff, 0, bfA);
                const bool last = (dh == 1) & (dw == 1);
                if (!last) loadB(wp + WSTRIDE_R, 0, bfA);
                computeChunk(aoff, 64, bfB);
                wp += WSTRIDE_R;
            }
        }
    }

    // ---- epilogue: out = inv_std*(acc - mean*wsum) + bias
#pragma unroll
    for (int j = 0; j < 4; ++j) {
        const int f     = waveN * 64 + j * 16 + lm;   // C/D: col = lane&15
        const float wsf = wsum[f];
        const float bsf = bias[f];
#pragma unroll
        for (int i = 0; i < 4; ++i) {
            const int mrow = m0 + waveM * 64 + i * 16 + lh * 8;  // mult of 8
            const int ob = mrow / L_PIX;
            const int op = mrow - ob * L_PIX;                     // op..op+7 in-image
            const float4 mu0 = *(const float4*)(meanArr   + mrow);
            const float4 mu1 = *(const float4*)(meanArr   + mrow + 4);
            const float4 is0 = *(const float4*)(invstdArr + mrow);
            const float4 is1 = *(const float4*)(invstdArr + mrow + 4);
            float* po = out + ((size_t)(ob * F_OUT + f)) * L_PIX + op;
            float4 o0, o1;
            o0.x = is0.x * (acc[i][j][0] - mu0.x * wsf) + bsf;
            o0.y = is0.y * (acc[i][j][1] - mu0.y * wsf) + bsf;
            o0.z = is0.z * (acc[i][j][2] - mu0.z * wsf) + bsf;
            o0.w = is0.w * (acc[i][j][3] - mu0.w * wsf) + bsf;
            o1.x = is1.x * (acc[i][j][4] - mu1.x * wsf) + bsf;
            o1.y = is1.y * (acc[i][j][5] - mu1.y * wsf) + bsf;
            o1.z = is1.z * (acc[i][j][6] - mu1.z * wsf) + bsf;
            o1.w = is1.w * (acc[i][j][7] - mu1.w * wsf) + bsf;
            *(float4*)(po)     = o0;
            *(float4*)(po + 4) = o1;
        }
    }
}

// ---------------------------------------------------------------------------
extern "C" void kernel_launch(void* const* d_in, const int* in_sizes, int n_in,
                              void* d_out, int out_size, void* d_ws, size_t ws_size,
                              hipStream_t stream) {
    const float* x    = (const float*)d_in[0];
    const float* w    = (const float*)d_in[1];
    const float* bias = (const float*)d_in[2];
    float* out = (float*)d_out;

    float*  meanArr   = (float*)d_ws;
    float*  invstdArr = meanArr   + M_TOT;
    float*  s1        = invstdArr + M_TOT;
    float*  s2        = s1        + M_TOT;
    float*  wsum      = s2        + M_TOT;
    __bf16* Wp        = (__bf16*)(wsum + F_OUT);   // 9*256*128 bf16, 16B-aligned

    k_prep_x<<<M_TOT / 256, 256, 0, stream>>>(x, s1, s2);
    k_stats <<<M_TOT / 256, 256, 0, stream>>>(s1, s2, meanArr, invstdArr);
    k_prep_w<<<F_OUT, 128, 0, stream>>>(w, Wp, wsum);

    k_gemm<<<M_TOT / BM, 256, 0, stream>>>(x, Wp, meanArr, invstdArr,
                                           wsum, bias, out);
}
